// RWKV_TinyAttn_12369505812815
// MI455X (gfx1250) — matrix operand order repaired and checked by reading, not compiled, not run on hardware
//
#include <hip/hip_runtime.h>
#include <hip/hip_bf16.h>
#include <math.h>

// ---------------------------------------------------------------------------
// Causal MHA block for MI455X (gfx1250, wave32, WMMA).
//  - bf16 WMMA (16x16x32) everywhere, f32 accumulate
//  - 128x128 block / 64x64 per-wave register tiles in the projections
//  - async global->LDS staging (ASYNCcnt) with double buffering
//  - explicit addrspace(3) fragment loads (ds_load, not flat_load)
//  - flash attention with block-shared K/V LDS tiles + lane-resident softmax
// ---------------------------------------------------------------------------

typedef __bf16 bf16;
typedef __attribute__((ext_vector_type(16))) __bf16 v16bf;
typedef __attribute__((ext_vector_type(8)))  __bf16 v8bf;
typedef __attribute__((ext_vector_type(8)))  float  v8f;

#define LDS_AS __attribute__((address_space(3)))

#define BB   4
#define TT   2048
#define CC   1024
#define DA   1024
#define NH   16
#define HD   64
#define D3   3072
#define BT   (BB * TT)

#define APAD 40   // LDS row stride (bf16) for 32-wide k slices (bank-conflict pad)
#define KPAD 72   // LDS row stride (bf16) for 64-wide K rows

// ----------------------------- WMMA helpers --------------------------------

__device__ __forceinline__ v8f wmma_bf16(v16bf a, v16bf b, v8f c) {
  return __builtin_amdgcn_wmma_f32_16x16x32_bf16(
      false, a, false, b, (short)0, c, false, false);
}

__device__ __forceinline__ v16bf cat8(v8bf lo, v8bf hi) {
  return __builtin_shufflevector(lo, hi, 0, 1, 2, 3, 4, 5, 6, 7,
                                         8, 9, 10, 11, 12, 13, 14, 15);
}

// ---- LDS access via explicit addrspace(3) (guaranteed ds_load/ds_store) ----

__device__ __forceinline__ unsigned lds_off(const void* p) {
  return (unsigned)(uintptr_t)p;   // generic LDS ptr carries AS3 offset in [31:0]
}

__device__ __forceinline__ v8bf lds_ld8(unsigned byteoff) {
  return *(const LDS_AS v8bf*)byteoff;         // ds_load_b128
}

__device__ __forceinline__ void lds_st16(unsigned byteoff, bf16 v) {
  *(LDS_AS bf16*)byteoff = v;                  // ds_store_b16
}

// 16 contiguous bf16 from LDS (two 16B ds loads)
__device__ __forceinline__ v16bf frag16_lds(unsigned off) {
  return cat8(lds_ld8(off), lds_ld8(off + 16));
}

// A-fragment (16x32) from LDS: row base byte offset; k = g*8 and 16+g*8.
__device__ __forceinline__ v16bf fragA_lds(unsigned rowoff, int g) {
  unsigned o = rowoff + g * 16;
  return cat8(lds_ld8(o), lds_ld8(o + 32));
}

// A-fragment (16x32) from global memory
__device__ __forceinline__ v16bf fragA_g(const bf16* rowp, int g) {
  const bf16* p = rowp + g * 8;
  return cat8(*(const v8bf*)p, *(const v8bf*)(p + 16));
}

// ----------------------------- async global->LDS ---------------------------

// One lane moves 16B; one wave instruction moves 512B. Tracked by ASYNCcnt.
__device__ __forceinline__ void async_cp16(unsigned loff, const bf16* g) {
  asm volatile("global_load_async_to_lds_b128 %0, %1, off"
               :: "v"(loff), "v"((unsigned long long)(uintptr_t)g)
               : "memory");
}

template <int N> __device__ __forceinline__ void wait_async() {
  asm volatile("s_wait_asynccnt %0" :: "i"(N) : "memory");
}

// ----------------------------- convert kernels -----------------------------

__global__ __launch_bounds__(256) void k_cvt_bf16(const float* __restrict__ src,
                                                  bf16* __restrict__ dst, int n) {
  int i = blockIdx.x * 256 + threadIdx.x;
  if (i < n) dst[i] = (bf16)src[i];
}

// dst[n*K + k] = (bf16) src[k*N + n]   (B operand becomes K-contiguous per column)
__global__ __launch_bounds__(256) void k_cvt_transpose(const float* __restrict__ src,
                                                       bf16* __restrict__ dst,
                                                       int K, int N) {
  int i = blockIdx.x * 256 + threadIdx.x;
  if (i >= K * N) return;
  int n = i / K;
  int k = i - n * K;
  dst[(size_t)n * K + k] = (bf16)src[(size_t)k * N + n];
}

// ----------------------------- staged GEMM core ----------------------------
// Block tile 128(M) x 128(N), 4 waves in 2x2 grid, 64x64 per wave.
// K-slices of 32 staged via async copies, double buffered.
// All LDS addressing is 32-bit byte offsets (aOff/bOff = buffer 0 base).

#define TILE_BYTES (128 * APAD * 2)

__device__ __forceinline__ void gemm_stage(const bf16* __restrict__ Ag,
                                           const bf16* __restrict__ Bg,
                                           int k0, int Kdim,
                                           unsigned aOff, unsigned bOff, int tid) {
  // A: 128 rows x 32 k = 512 x 16B chunks; 4 per thread. Same for B.
#pragma unroll
  for (int u = 0; u < 4; ++u) {
    int c = tid * 4 + u;
    int row = c >> 2, cc = c & 3;
    async_cp16(aOff + (unsigned)(row * APAD + cc * 8) * 2,
               Ag + (size_t)row * Kdim + k0 + cc * 8);
  }
#pragma unroll
  for (int u = 0; u < 4; ++u) {
    int c = tid * 4 + u;
    int row = c >> 2, cc = c & 3;
    async_cp16(bOff + (unsigned)(row * APAD + cc * 8) * 2,
               Bg + (size_t)row * Kdim + k0 + cc * 8);
  }
}

__device__ __forceinline__ void gemm_mma(const bf16* __restrict__ Ag,
                                         const bf16* __restrict__ Bg,
                                         int Kdim, unsigned aOff, unsigned bOff,
                                         v8f acc[4][4],
                                         int tid, int ln, int g, int wm, int wn) {
  const int nsteps = Kdim / 32;
  gemm_stage(Ag, Bg, 0, Kdim, aOff, bOff, tid);
  for (int i = 0; i < nsteps; ++i) {
    if (i + 1 < nsteps) {
      unsigned nbuf = (unsigned)((i + 1) & 1) * TILE_BYTES;
      gemm_stage(Ag, Bg, (i + 1) * 32, Kdim, aOff + nbuf, bOff + nbuf, tid);
      wait_async<8>();     // batch i landed (async loads complete in order)
    } else {
      wait_async<0>();
    }
    __syncthreads();
    unsigned cbuf = (unsigned)(i & 1) * TILE_BYTES;
    unsigned aB = aOff + cbuf;
    unsigned bB = bOff + cbuf;
    v16bf af[4], bfr[4];
#pragma unroll
    for (int at = 0; at < 4; ++at)
      af[at] = fragA_lds(aB + (unsigned)((wm * 64 + at * 16 + ln) * APAD) * 2, g);
#pragma unroll
    for (int bt = 0; bt < 4; ++bt)
      bfr[bt] = fragA_lds(bB + (unsigned)((wn * 64 + bt * 16 + ln) * APAD) * 2, g);
#pragma unroll
    for (int at = 0; at < 4; ++at)
#pragma unroll
      for (int bt = 0; bt < 4; ++bt)
        acc[at][bt] = wmma_bf16(af[at], bfr[bt], acc[at][bt]);
    __syncthreads();       // reads done before buffer is overwritten next iter
  }
}

// ----------------------------- QKV projection ------------------------------

__global__ __launch_bounds__(128) void k_gemm_qkv(const bf16* __restrict__ xb,
                                                  const bf16* __restrict__ wqt,
                                                  const float* __restrict__ bqkv,
                                                  bf16* __restrict__ Qw,
                                                  bf16* __restrict__ Kw,
                                                  bf16* __restrict__ Vt) {
  __shared__ bf16 sA[2][128 * APAD];
  __shared__ bf16 sB[2][128 * APAD];

  const int tid  = threadIdx.x;
  const int lane = tid & 31;
  const int w    = tid >> 5;
  const int ln   = lane & 15;
  const int g    = lane >> 4;
  const int wm   = w >> 1, wn = w & 1;
  const int m0   = blockIdx.x * 128;
  const int n0   = blockIdx.y * 128;

  v8f acc[4][4] = {};
  gemm_mma(xb + (size_t)m0 * CC, wqt + (size_t)n0 * CC, CC,
           lds_off(&sA[0][0]), lds_off(&sB[0][0]), acc, tid, ln, g, wm, wn);

#pragma unroll
  for (int at = 0; at < 4; ++at)
#pragma unroll
    for (int bt = 0; bt < 4; ++bt)
#pragma unroll
      for (int v = 0; v < 8; ++v) {
        int row = m0 + wm * 64 + at * 16 + v + 8 * g;
        int col = n0 + wn * 64 + bt * 16 + ln;
        float val = acc[at][bt][v] + bqkv[col];
        int sec = col >> 10;          // 0:Q 1:K 2:V
        int a   = col & 1023;
        int h   = a >> 6;
        int d   = a & 63;
        int b   = row >> 11;
        int tt  = row & 2047;
        int bh  = b * NH + h;
        if (sec == 0) {
          Qw[((size_t)bh * TT + tt) * HD + d] = (bf16)(val * 0.125f);
        } else if (sec == 1) {
          Kw[((size_t)bh * TT + tt) * HD + d] = (bf16)val;
        } else {
          Vt[((size_t)bh * HD + d) * TT + tt] = (bf16)val;
        }
      }
}

// ----------------------------- flash attention -----------------------------
// Block = (bh, 64 q rows), 4 waves x 16 q rows. K/V tiles (32 kv wide) are
// async-staged into LDS once per block and consumed by all 4 waves.

#define KTILE_BYTES (32 * KPAD * 2)
#define VTILE_BYTES (64 * APAD * 2)

__global__ __launch_bounds__(128) void k_flash(const bf16* __restrict__ Qw,
                                               const bf16* __restrict__ Kw,
                                               const bf16* __restrict__ Vt,
                                               bf16* __restrict__ AO) {
  __shared__ bf16 sK[2][32 * KPAD];   // 32 kv rows x 64 dims
  __shared__ bf16 sV[2][64 * APAD];   // 64 dims x 32 kv
  __shared__ bf16 sP[4][16][32];      // per-wave P staging

  const int tid  = threadIdx.x;
  const int lane = tid & 31;
  const int w    = tid >> 5;
  const int ln   = lane & 15;
  const int g    = lane >> 4;
  const int bh   = blockIdx.y;
  const int b    = bh >> 4;
  const int h    = bh & 15;
  const int q0   = blockIdx.x * 64 + w * 16;

  const bf16* Qbase = Qw + (size_t)bh * TT * HD;
  const bf16* Kbase = Kw + (size_t)bh * TT * HD;
  const bf16* Vbase = Vt + (size_t)bh * HD * TT;

  const unsigned kOff0 = lds_off(&sK[0][0]);
  const unsigned vOff0 = lds_off(&sV[0][0]);
  const unsigned pOffW = lds_off(&sP[w][0][0]);

  const bf16* qrow = Qbase + (size_t)(q0 + ln) * HD;
  v16bf qa0 = fragA_g(qrow, g);
  v16bf qa1 = fragA_g(qrow + 32, g);

  v8f   o[4] = {};
  float mr[8], lr[8];
#pragma unroll
  for (int v = 0; v < 8; ++v) { mr[v] = -1e30f; lr[v] = 0.0f; }

  const int nb = (blockIdx.x * 64 + 64) / 32;   // causal bound, uniform in block

  auto stage_kv = [&](int ib, int buf) {
    int kv0 = ib * 32;
    unsigned kO = kOff0 + (unsigned)buf * KTILE_BYTES;
    unsigned vO = vOff0 + (unsigned)buf * VTILE_BYTES;
#pragma unroll
    for (int u = 0; u < 2; ++u) {               // K: 256 chunks of 16B
      int c = tid * 2 + u;
      int r = c >> 3, cc = c & 7;
      async_cp16(kO + (unsigned)(r * KPAD + cc * 8) * 2,
                 Kbase + (size_t)(kv0 + r) * HD + cc * 8);
    }
#pragma unroll
    for (int u = 0; u < 2; ++u) {               // V^T: 256 chunks of 16B
      int c = tid * 2 + u;
      int d = c >> 2, cc = c & 3;
      async_cp16(vO + (unsigned)(d * APAD + cc * 8) * 2,
                 Vbase + (size_t)d * TT + kv0 + cc * 8);
    }
  };

  stage_kv(0, 0);
  for (int ib = 0; ib < nb; ++ib) {
    const int kv0 = ib * 32;
    if (ib + 1 < nb) { stage_kv(ib + 1, (ib + 1) & 1); wait_async<4>(); }
    else             { wait_async<0>(); }
    __syncthreads();
    const unsigned kB = kOff0 + (unsigned)(ib & 1) * KTILE_BYTES;
    const unsigned vB = vOff0 + (unsigned)(ib & 1) * VTILE_BYTES;

    // ---- scores: two 16x16 tiles over the 32-wide kv block ----
    v8f s0 = {}, s1 = {};
    {
      unsigned kc0 = kB + (unsigned)(ln * KPAD) * 2;
      s0 = wmma_bf16(qa0, fragA_lds(kc0, g), s0);
      s0 = wmma_bf16(qa1, fragA_lds(kc0 + 64, g), s0);
      unsigned kc1 = kB + (unsigned)((16 + ln) * KPAD) * 2;
      s1 = wmma_bf16(qa0, fragA_lds(kc1, g), s1);
      s1 = wmma_bf16(qa1, fragA_lds(kc1 + 64, g), s1);
    }

    // ---- causal mask + online softmax (lane-resident m/l) ----
    float alpha[8];
#pragma unroll
    for (int v = 0; v < 8; ++v) {
      int row = q0 + v + 8 * g;
      float a0 = ((kv0 + ln)      <= row) ? s0[v] : -1e30f;
      float a1 = ((kv0 + 16 + ln) <= row) ? s1[v] : -1e30f;
      float x = fmaxf(a0, a1);
#pragma unroll
      for (int off = 1; off < 16; off <<= 1)
        x = fmaxf(x, __shfl_xor(x, off, 16));
      float mn = fmaxf(mr[v], x);
      float al = __expf(mr[v] - mn);
      float p0 = __expf(a0 - mn);
      float p1 = __expf(a1 - mn);
      float ls = p0 + p1;
#pragma unroll
      for (int off = 1; off < 16; off <<= 1)
        ls += __shfl_xor(ls, off, 16);
      lr[v] = lr[v] * al + ls;
      mr[v] = mn;
      alpha[v] = al;
      unsigned pr = pOffW + (unsigned)((v + 8 * g) * 32) * 2;
      lds_st16(pr + (unsigned)ln * 2, (bf16)p0);
      lds_st16(pr + (unsigned)(16 + ln) * 2, (bf16)p1);
    }

    asm volatile("s_wait_dscnt 0" ::: "memory");  // cross-lane P relayout fence

#pragma unroll
    for (int t = 0; t < 4; ++t)
#pragma unroll
      for (int v = 0; v < 8; ++v) o[t][v] *= alpha[v];

    v16bf pa = fragA_lds(pOffW + (unsigned)(ln * 32) * 2, g);
#pragma unroll
    for (int t = 0; t < 4; ++t) {
      unsigned vc = vB + (unsigned)((t * 16 + ln) * APAD) * 2;
      o[t] = wmma_bf16(pa, fragA_lds(vc, g), o[t]);
    }
    __syncthreads();    // K/V reads done before next staging overwrites
  }

  float invl[8];
#pragma unroll
  for (int v = 0; v < 8; ++v) invl[v] = 1.0f / lr[v];
#pragma unroll
  for (int t = 0; t < 4; ++t)
#pragma unroll
    for (int v = 0; v < 8; ++v) {
      int row = q0 + v + 8 * g;
      AO[((size_t)(b * TT + row)) * DA + h * HD + t * 16 + ln] =
          (bf16)(o[t][v] * invl[v]);
    }
}

// ----------------------------- output projection ---------------------------

__global__ __launch_bounds__(128) void k_gemm_out(const bf16* __restrict__ AO,
                                                  const bf16* __restrict__ wot,
                                                  const float* __restrict__ bout,
                                                  float* __restrict__ out) {
  __shared__ bf16 sA[2][128 * APAD];
  __shared__ bf16 sB[2][128 * APAD];

  const int tid  = threadIdx.x;
  const int lane = tid & 31;
  const int w    = tid >> 5;
  const int ln   = lane & 15;
  const int g    = lane >> 4;
  const int wm   = w >> 1, wn = w & 1;
  const int m0   = blockIdx.x * 128;
  const int n0   = blockIdx.y * 128;

  v8f acc[4][4] = {};
  gemm_mma(AO + (size_t)m0 * DA, wot + (size_t)n0 * DA, DA,
           lds_off(&sA[0][0]), lds_off(&sB[0][0]), acc, tid, ln, g, wm, wn);

#pragma unroll
  for (int at = 0; at < 4; ++at)
#pragma unroll
    for (int bt = 0; bt < 4; ++bt)
#pragma unroll
      for (int v = 0; v < 8; ++v) {
        int row = m0 + wm * 64 + at * 16 + v + 8 * g;
        int col = n0 + wn * 64 + bt * 16 + ln;
        out[(size_t)row * CC + col] = acc[at][bt][v] + bout[col];
      }
}

// ----------------------------- launcher ------------------------------------

extern "C" void kernel_launch(void* const* d_in, const int* in_sizes, int n_in,
                              void* d_out, int out_size, void* d_ws, size_t ws_size,
                              hipStream_t stream) {
  (void)in_sizes; (void)n_in; (void)out_size; (void)ws_size;

  const float* x    = (const float*)d_in[0];
  // d_in[1] = causal mask (recomputed from indices)
  const float* Wqkv = (const float*)d_in[2];
  const float* bqkv = (const float*)d_in[3];
  const float* Wout = (const float*)d_in[4];
  const float* bout = (const float*)d_in[5];
  float*       out  = (float*)d_out;

  char* ws = (char*)d_ws;
  bf16* xb  = (bf16*)(ws + 0);          // 16 MB  x bf16 [8192][1024]
  bf16* wqt = (bf16*)(ws + 16777216);   //  6 MB  W_qkv^T bf16 [3072][1024]
  bf16* wot = (bf16*)(ws + 23068672);   //  2 MB  W_out^T bf16 [1024][1024]
  bf16* Qw  = (bf16*)(ws + 25165824);   // 16 MB  Q [64][2048][64] (pre-scaled)
  bf16* Kw  = (bf16*)(ws + 41943040);   // 16 MB  K [64][2048][64]
  bf16* Vt  = (bf16*)(ws + 58720256);   // 16 MB  V^T [64][64][2048]
  bf16* AO  = (bf16*)(ws + 75497472);   // 16 MB  attn out [4][2048][1024]

  k_cvt_bf16<<<(BT * CC) / 256, 256, 0, stream>>>(x, xb, BT * CC);
  k_cvt_transpose<<<(D3 * CC) / 256, 256, 0, stream>>>(Wqkv, wqt, CC, D3);
  k_cvt_transpose<<<(DA * CC) / 256, 256, 0, stream>>>(Wout, wot, DA, CC);

  k_gemm_qkv<<<dim3(BT / 128, D3 / 128), 128, 0, stream>>>(xb, wqt, bqkv, Qw, Kw, Vt);
  k_flash<<<dim3(TT / 64, BB * NH), 128, 0, stream>>>(Qw, Kw, Vt, AO);
  k_gemm_out<<<dim3(BT / 128, CC / 128), 128, 0, stream>>>(AO, wot, bout, out);
}
